// MultiScaleAttention_40175124087082
// MI455X (gfx1250) — matrix operand order repaired and checked by reading, not compiled, not run on hardware
//
#include <hip/hip_runtime.h>

// ---------------------------------------------------------------------------
// MultiScaleAttention for MI455X (gfx1250).
// All matmuls on v_wmma_f32_16x16x32_bf16 (fp32 accum), flash softmax in fp32.
// GEMMs + attention: double-buffered operand tiles staged in LDS with gfx1250
// async-to-LDS loads (ASYNCcnt), A fragments ping-ponged in registers.
// ---------------------------------------------------------------------------

typedef __attribute__((ext_vector_type(8)))  __bf16 v8bf;
typedef __attribute__((ext_vector_type(16))) __bf16 v16bf;
typedef __attribute__((ext_vector_type(8)))  float  v8f;
typedef int v4i_vs __attribute__((vector_size(4 * sizeof(int))));  // b128 payload

#define D_HID   1024
#define L_SEQ   2048
#define N_BATCH 2
#define N_HEADS 16
#define D_HEAD  64
#define M_ROWS  (N_BATCH * L_SEQ)   // 4096
#define N_QKV   (3 * D_HID)         // 3072

#if defined(__has_builtin)
#if __has_builtin(__builtin_amdgcn_global_load_async_to_lds_b128) && \
    __has_builtin(__builtin_amdgcn_s_wait_asynccnt)
#define HAVE_ASYNC_LDS 1
#endif
#endif
#ifndef HAVE_ASYNC_LDS
#define HAVE_ASYNC_LDS 0
#endif

static __device__ __forceinline__ v16bf pack16(v8bf lo, v8bf hi) {
  return __builtin_shufflevector(lo, hi, 0,1,2,3,4,5,6,7,8,9,10,11,12,13,14,15);
}
static __device__ __forceinline__ v8f wmma_bf16(v16bf a, v16bf b, v8f c) {
  return __builtin_amdgcn_wmma_f32_16x16x32_bf16(false, a, false, b, (short)0, c,
                                                 false, false);
}
// Row reductions across the 16 lanes of a half-wave (wave32; xor masks < 16
// never leave the half, so lanes 0-15 and 16-31 reduce independently).
static __device__ __forceinline__ float redmax16(float v) {
  v = fmaxf(v, __shfl_xor(v, 1, 32));
  v = fmaxf(v, __shfl_xor(v, 2, 32));
  v = fmaxf(v, __shfl_xor(v, 4, 32));
  v = fmaxf(v, __shfl_xor(v, 8, 32));
  return v;
}
static __device__ __forceinline__ float redsum16(float v) {
  v += __shfl_xor(v, 1, 32);
  v += __shfl_xor(v, 2, 32);
  v += __shfl_xor(v, 4, 32);
  v += __shfl_xor(v, 8, 32);
  return v;
}

// 16B global -> LDS copy: async (ASYNCcnt) when the gfx1250 builtin is
// available, otherwise synchronous global->reg->ds fallback.
// Builtin signature (probe-verified): (v4i AS1*, v4i AS3*, imm offset, imm cpol).
static __device__ __forceinline__ void cp16_g2l(const __bf16* g, __bf16* l) {
#if HAVE_ASYNC_LDS
  __builtin_amdgcn_global_load_async_to_lds_b128(
      (__attribute__((address_space(1))) v4i_vs*)(uintptr_t)g,
      (__attribute__((address_space(3))) v4i_vs*)(uint32_t)(uintptr_t)l,
      0, 0);
#else
  *(v8bf*)l = *(const v8bf*)g;
#endif
}
template <int N>
static __device__ __forceinline__ void async_wait_le() {
#if HAVE_ASYNC_LDS
  __builtin_amdgcn_s_wait_asynccnt(N);   // constant immediate required
#endif
}

// --------------------------- fp32 -> bf16 convert ---------------------------
__global__ __launch_bounds__(256) void cvt_bf16_kernel(const float* __restrict__ in,
                                                       __bf16* __restrict__ out, int n) {
  int i = (blockIdx.x * 256 + threadIdx.x) * 4;
  if (i + 3 < n) {
    float4 v = *(const float4*)(in + i);
    out[i + 0] = (__bf16)v.x;
    out[i + 1] = (__bf16)v.y;
    out[i + 2] = (__bf16)v.z;
    out[i + 3] = (__bf16)v.w;
  }
}

// ------------------- fp32 [K][N] -> bf16 transposed [N][K] ------------------
__global__ __launch_bounds__(256) void transpose_cvt_kernel(const float* __restrict__ in,
                                                            __bf16* __restrict__ out,
                                                            int K, int N) {
  __shared__ float tile[32][33];
  int kb = blockIdx.y * 32, nb = blockIdx.x * 32;
  int tx = threadIdx.x, ty = threadIdx.y;           // block (32, 8)
#pragma unroll
  for (int r = ty; r < 32; r += 8)
    tile[r][tx] = in[(size_t)(kb + r) * N + nb + tx];
  __syncthreads();
#pragma unroll
  for (int r = ty; r < 32; r += 8)
    out[(size_t)(nb + r) * K + kb + tx] = (__bf16)tile[tx][r];
}

// ---------------------- shared GEMM core (LDS-staged B) ----------------------
// Block 128x64 tile, 4 waves; wave = 32 rows x 64 cols. B tiles (64n x 64k)
// double-buffered in LDS via async copies; A fragments ping-ponged in VGPRs.
static __device__ __forceinline__ void gemm_core_lds(
    const __bf16* __restrict__ aMat, const __bf16* __restrict__ wT,
    __bf16 (*Bl)[64][64], int nbase, int mbase, int tid, v8f (&acc)[8]) {
  const int lane = tid & 31;
  const int half = lane >> 4, lm = lane & 15;
  const __bf16* a0p = aMat + (size_t)(mbase + lm) * D_HID + 8 * half;
  const __bf16* a1p = aMat + (size_t)(mbase + 16 + lm) * D_HID + 8 * half;

  // Cooperative B-tile loader: 64 rows x 128B = 512 x 16B chunks / 128 threads.
  auto load_btile = [&](int k0, int buf) {
#pragma unroll
    for (int i = 0; i < 4; i++) {
      int c = tid + 128 * i;          // 0..511
      int r = c >> 3;                 // n-local row 0..63
      int koff = (c & 7) * 8;         // bf16 col, step 8 (16B)
      cp16_g2l(wT + (size_t)(nbase + r) * D_HID + k0 + koff, &Bl[buf][r][koff]);
    }
  };
  // A fragments for one 64-k tile: [kstep][mt].
  auto load_afrags = [&](int k0, v16bf (&A)[2][2]) {
#pragma unroll
    for (int ks = 0; ks < 2; ks++) {
      A[ks][0] = pack16(*(const v8bf*)(a0p + k0 + ks * 32),
                        *(const v8bf*)(a0p + k0 + ks * 32 + 16));
      A[ks][1] = pack16(*(const v8bf*)(a1p + k0 + ks * 32),
                        *(const v8bf*)(a1p + k0 + ks * 32 + 16));
    }
  };
  auto tile_mma = [&](int buf, const v16bf (&A)[2][2]) {
#pragma unroll
    for (int ks = 0; ks < 2; ks++)
#pragma unroll
      for (int nt = 0; nt < 4; nt++) {
        v16bf bfrag = pack16(*(const v8bf*)&Bl[buf][nt * 16 + lm][ks * 32 + 8 * half],
                             *(const v8bf*)&Bl[buf][nt * 16 + lm][ks * 32 + 16 + 8 * half]);
#pragma unroll
        for (int mt = 0; mt < 2; mt++)
          acc[mt * 4 + nt] = wmma_bf16(A[ks][mt], bfrag, acc[mt * 4 + nt]);
      }
  };

  const int NKT = D_HID / 64;   // 16 k-tiles (even)
  v16bf Aa[2][2], Ab[2][2];
  load_btile(0, 0);
  load_afrags(0, Aa);
  for (int t = 0; t < NKT; t += 2) {
    // tile t (buffer 0 / Aa): stage tile t+1 while computing
    load_btile((t + 1) * 64, 1);
    load_afrags((t + 1) * 64, Ab);
    async_wait_le<4>();               // tile-t async transfers complete
    __syncthreads();
    tile_mma(0, Aa);
    __syncthreads();                  // all waves done with buffer 0
    // tile t+1 (buffer 1 / Ab): stage tile t+2 while computing
    if (t + 2 < NKT) {
      load_btile((t + 2) * 64, 0);
      load_afrags((t + 2) * 64, Aa);
      async_wait_le<4>();
    } else {
      async_wait_le<0>();
    }
    __syncthreads();
    tile_mma(1, Ab);
    __syncthreads();
  }
}

// ------------------------------- QKV GEMM -----------------------------------
// C[4096,3072] = Xb @ Wqkv + b -> Q,K [B,H,L,Dh], V^T [B,H,Dh,L] (bf16).
__global__ __launch_bounds__(128) void qkv_gemm_kernel(const __bf16* __restrict__ xb,
                                                       const __bf16* __restrict__ wT,
                                                       const float* __restrict__ bias,
                                                       __bf16* __restrict__ Qb,
                                                       __bf16* __restrict__ Kb,
                                                       __bf16* __restrict__ Vt) {
  __shared__ __attribute__((aligned(32))) __bf16 Bl[2][64][64];
  const int tid = threadIdx.x;
  const int w = tid >> 5, lane = tid & 31;
  const int half = lane >> 4, lm = lane & 15;
  const int nbase = blockIdx.x * 64;
  const int mbase = blockIdx.y * 128 + w * 32;

  v8f acc[8] = {v8f{}, v8f{}, v8f{}, v8f{}, v8f{}, v8f{}, v8f{}, v8f{}};
  gemm_core_lds(xb, wT, Bl, nbase, mbase, tid, acc);

  // Epilogue: C layout -> row 8*half+j, col lane%16.
#pragma unroll
  for (int mt = 0; mt < 2; mt++) {
#pragma unroll
    for (int nt = 0; nt < 4; nt++) {
      int n = nbase + nt * 16 + lm;
      float bv = bias[n];
      int region = n >> 10;            // 0=Q, 1=K, 2=V
      int d = n & (D_HID - 1);
      int hh = d >> 6, dh = d & 63;
#pragma unroll
      for (int j = 0; j < 8; j++) {
        int row = mbase + mt * 16 + 8 * half + j;   // b*L + l
        int b = row >> 11, l = row & (L_SEQ - 1);
        float val = acc[mt * 4 + nt][j] + bv;
        size_t bh = (size_t)(b * N_HEADS + hh);
        if (region == 0)      Qb[(bh * L_SEQ + l) * D_HEAD + dh] = (__bf16)(val * 0.125f);
        else if (region == 1) Kb[(bh * L_SEQ + l) * D_HEAD + dh] = (__bf16)val;
        else                  Vt[(bh * D_HEAD + dh) * L_SEQ + l] = (__bf16)val;
      }
    }
  }
}

// ----------------------------- Flash attention ------------------------------
// grid (B*H, L/64); block 128 = 4 waves, 64 q rows. K/V tiles of 64 keys are
// double-buffered in LDS (async global->LDS); all 4 waves share each tile.
__global__ __launch_bounds__(128) void attn_kernel(const __bf16* __restrict__ Qb,
                                                   const __bf16* __restrict__ Kb,
                                                   const __bf16* __restrict__ Vt,
                                                   __bf16* __restrict__ aout) {
  __shared__ __attribute__((aligned(32))) __bf16 Kl[2][64][64];   // [buf][key][dh]
  __shared__ __attribute__((aligned(32))) __bf16 Vl[2][64][64];   // [buf][dh][kv]
  __shared__ __attribute__((aligned(32))) __bf16 plds[4][16][64]; // per-wave P tile

  const int bh = blockIdx.x;                     // 0..31
  const int q0 = blockIdx.y * 64;
  const int tid = threadIdx.x;
  const int w = tid >> 5, lane = tid & 31;
  const int half = lane >> 4, lm = lane & 15;
  const int b = bh >> 4, h = bh & 15;
  const __bf16* Qh = Qb + (size_t)bh * L_SEQ * D_HEAD;
  const __bf16* Kh = Kb + (size_t)bh * L_SEQ * D_HEAD;
  const __bf16* Vh = Vt + (size_t)bh * D_HEAD * L_SEQ;

  // Q fragments for this wave's 16 rows (A-frag row = lane%16), kept in VGPRs.
  const int qa = q0 + w * 16 + lm;
  v16bf qf[2];
#pragma unroll
  for (int ks = 0; ks < 2; ks++) {
    v8bf lo = *(const v8bf*)(Qh + (size_t)qa * D_HEAD + ks * 32 + 8 * half);
    v8bf hi = *(const v8bf*)(Qh + (size_t)qa * D_HEAD + ks * 32 + 16 + 8 * half);
    qf[ks] = pack16(lo, hi);
  }

  v8f o[4] = {v8f{}, v8f{}, v8f{}, v8f{}};
  float rm[8], rs[8];
#pragma unroll
  for (int j = 0; j < 8; j++) { rm[j] = -INFINITY; rs[j] = 0.0f; }

  // Cooperative tile loader: 128 threads x 4 chunks x 16B = 8KB per tile.
  auto load_tiles = [&](int kv, int buf) {
#pragma unroll
    for (int i = 0; i < 4; i++) {
      int c = tid + 128 * i;          // 0..511
      int r = c >> 3;                 // row 0..63
      int col = (c & 7) * 8;          // bf16 col, step 8 (16B)
      cp16_g2l(Kh + (size_t)(kv + r) * D_HEAD + col, &Kl[buf][r][col]);
    }
#pragma unroll
    for (int i = 0; i < 4; i++) {
      int c = tid + 128 * i;
      int r = c >> 3;                 // dh row
      int col = (c & 7) * 8;          // kv col
      cp16_g2l(Vh + (size_t)r * L_SEQ + kv + col, &Vl[buf][r][col]);
    }
  };

  const int NT = L_SEQ / 64;          // 32 kv tiles
  load_tiles(0, 0);
  for (int t = 0; t < NT; t++) {
    if (t + 1 < NT) {
      load_tiles((t + 1) * 64, (t + 1) & 1);
      async_wait_le<8>();             // tile-t transfers (issued first) complete
    } else {
      async_wait_le<0>();             // drain
    }
    __syncthreads();                  // all waves' tile-t data visible

    const __bf16 (*Kc)[64] = Kl[t & 1];
    const __bf16 (*Vc)[64] = Vl[t & 1];

    // S = Q @ K^T : 16 q rows x 64 keys (scale folded into Q).
    v8f s[4] = {v8f{}, v8f{}, v8f{}, v8f{}};
#pragma unroll
    for (int ks = 0; ks < 2; ks++)
#pragma unroll
      for (int nt = 0; nt < 4; nt++) {
        v16bf bk = pack16(*(const v8bf*)&Kc[nt * 16 + lm][ks * 32 + 8 * half],
                          *(const v8bf*)&Kc[nt * 16 + lm][ks * 32 + 16 + 8 * half]);
        s[nt] = wmma_bf16(qf[ks], bk, s[nt]);
      }

    // Online softmax (fp32). Value j lives on row 8*half+j.
    float p[4][8], alpha[8];
#pragma unroll
    for (int j = 0; j < 8; j++) {
      float mx = fmaxf(fmaxf(s[0][j], s[1][j]), fmaxf(s[2][j], s[3][j]));
      mx = redmax16(mx);
      float nm = fmaxf(rm[j], mx);
      alpha[j] = __expf(rm[j] - nm);
      float tsum = 0.0f;
#pragma unroll
      for (int nt = 0; nt < 4; nt++) {
        p[nt][j] = __expf(s[nt][j] - nm);
        tsum += p[nt][j];
      }
      tsum = redsum16(tsum);
      rs[j] = rs[j] * alpha[j] + tsum;
      rm[j] = nm;
    }
#pragma unroll
    for (int nt = 0; nt < 4; nt++)
#pragma unroll
      for (int j = 0; j < 8; j++) o[nt][j] *= alpha[j];

    // Re-layout P (C-format -> A-fragment) through wave-private LDS.
    // LDS is in-order per wave (no cross-wave sharing here) -> compiler fence only.
#pragma unroll
    for (int j = 0; j < 8; j++)
#pragma unroll
      for (int nt = 0; nt < 4; nt++)
        plds[w][8 * half + j][nt * 16 + lm] = (__bf16)p[nt][j];
    asm volatile("" ::: "memory");
    v16bf pa[2];
#pragma unroll
    for (int ks = 0; ks < 2; ks++) {
      v8bf plo = *(const v8bf*)&plds[w][lm][ks * 32 + 8 * half];
      v8bf phi = *(const v8bf*)&plds[w][lm][ks * 32 + 16 + 8 * half];
      pa[ks] = pack16(plo, phi);
    }

    // O += P @ V (V^T tile: k runs contiguously over kv within a dh row).
#pragma unroll
    for (int nt = 0; nt < 4; nt++)
#pragma unroll
      for (int ks = 0; ks < 2; ks++) {
        v16bf bv = pack16(*(const v8bf*)&Vc[nt * 16 + lm][ks * 32 + 8 * half],
                          *(const v8bf*)&Vc[nt * 16 + lm][ks * 32 + 16 + 8 * half]);
        o[nt] = wmma_bf16(pa[ks], bv, o[nt]);
      }

    __syncthreads();                  // reads done before buffer reuse
  }

  float inv[8];
#pragma unroll
  for (int j = 0; j < 8; j++) inv[j] = 1.0f / rs[j];
#pragma unroll
  for (int nt = 0; nt < 4; nt++) {
#pragma unroll
    for (int j = 0; j < 8; j++) {
      int m = 8 * half + j;
      int qrow = q0 + w * 16 + m;
      aout[((size_t)b * L_SEQ + qrow) * D_HID + h * D_HEAD + nt * 16 + lm] =
          (__bf16)(o[nt][j] * inv[j]);
    }
  }
}

// ------------------------------- Proj GEMM ----------------------------------
__global__ __launch_bounds__(128) void proj_gemm_kernel(const __bf16* __restrict__ ab,
                                                        const __bf16* __restrict__ wT,
                                                        const float* __restrict__ bias,
                                                        float* __restrict__ out) {
  __shared__ __attribute__((aligned(32))) __bf16 Bl[2][64][64];
  const int tid = threadIdx.x;
  const int w = tid >> 5, lane = tid & 31;
  const int half = lane >> 4, lm = lane & 15;
  const int nbase = blockIdx.x * 64;
  const int mbase = blockIdx.y * 128 + w * 32;

  v8f acc[8] = {v8f{}, v8f{}, v8f{}, v8f{}, v8f{}, v8f{}, v8f{}, v8f{}};
  gemm_core_lds(ab, wT, Bl, nbase, mbase, tid, acc);

#pragma unroll
  for (int mt = 0; mt < 2; mt++) {
#pragma unroll
    for (int nt = 0; nt < 4; nt++) {
      int n = nbase + nt * 16 + lm;
      float bv = bias[n];
#pragma unroll
      for (int j = 0; j < 8; j++) {
        int row = mbase + mt * 16 + 8 * half + j;
        out[(size_t)row * D_HID + n] = acc[mt * 4 + nt][j] + bv;
      }
    }
  }
}

// ------------------------------- launcher -----------------------------------
extern "C" void kernel_launch(void* const* d_in, const int* in_sizes, int n_in,
                              void* d_out, int out_size, void* d_ws, size_t ws_size,
                              hipStream_t stream) {
  const float* x      = (const float*)d_in[0];   // [2,2048,1024]
  const float* w_qkv  = (const float*)d_in[1];   // [1024,3072]
  const float* b_qkv  = (const float*)d_in[2];   // [3072]
  const float* w_proj = (const float*)d_in[3];   // [1024,1024]
  const float* b_proj = (const float*)d_in[4];   // [1024]
  float* out = (float*)d_out;

  char* ws = (char*)d_ws;
  __bf16* xb     = (__bf16*)(ws);                          //  8 MiB
  __bf16* wqkvT  = (__bf16*)(ws + ((size_t)8  << 20));     //  6 MiB
  __bf16* wprojT = (__bf16*)(ws + ((size_t)14 << 20));     //  2 MiB
  __bf16* Qb     = (__bf16*)(ws + ((size_t)16 << 20));     //  8 MiB
  __bf16* Kb     = (__bf16*)(ws + ((size_t)24 << 20));     //  8 MiB
  __bf16* Vt     = (__bf16*)(ws + ((size_t)32 << 20));     //  8 MiB
  __bf16* aoutb  = (__bf16*)(ws + ((size_t)40 << 20));     //  8 MiB

  const int nx = M_ROWS * D_HID;  // 4194304
  cvt_bf16_kernel<<<nx / 4 / 256, 256, 0, stream>>>(x, xb, nx);
  transpose_cvt_kernel<<<dim3(N_QKV / 32, D_HID / 32), dim3(32, 8), 0, stream>>>(
      w_qkv, wqkvT, D_HID, N_QKV);
  transpose_cvt_kernel<<<dim3(D_HID / 32, D_HID / 32), dim3(32, 8), 0, stream>>>(
      w_proj, wprojT, D_HID, D_HID);

  qkv_gemm_kernel<<<dim3(N_QKV / 64, M_ROWS / 128), 128, 0, stream>>>(
      xb, wqkvT, b_qkv, Qb, Kb, Vt);

  attn_kernel<<<dim3(N_BATCH * N_HEADS, L_SEQ / 64), 128, 0, stream>>>(
      Qb, Kb, Vt, aoutb);

  proj_gemm_kernel<<<dim3(D_HID / 64, M_ROWS / 128), 128, 0, stream>>>(
      aoutb, wprojT, b_proj, out);
}
